// Qwen3Attention_44633300140490
// MI455X (gfx1250) — compile-verified, matrix operand order checked by reading, not run on hardware
//
#include <hip/hip_runtime.h>
#include <hip/hip_bf16.h>

#define DEV_INLINE __device__ __forceinline__

typedef __attribute__((ext_vector_type(16))) _Float16 v16h;
typedef __attribute__((ext_vector_type(8)))  _Float16 v8h;
typedef __attribute__((ext_vector_type(8)))  float    v8f;

namespace {
constexpr int   B_   = 2;
constexpr int   S_   = 1024;
constexpr int   HID_ = 2048;
constexpr int   NH_  = 16;
constexpr int   NKV_ = 8;
constexpr int   D_   = 128;
constexpr float EPS_ = 0.0001f / 65535.0f;

// GEMM tiling: 256 threads = 8 waves; block tile 128x64, wave tile 32x32 (2x2 WMMA 16x16x32)
constexpr int BM = 128, BN = 64, KT = 32, KTP = 48; // KTP padded so 32B v16h LDS loads stay aligned
}

// ---------------- reduction helpers ----------------
DEV_INLINE float opApply(float a, float b, int op) {
  return op == 0 ? fminf(a, b) : (op == 1 ? fmaxf(a, b) : a + b);
}

DEV_INLINE float blockReduce(float v, int op, float* sred) {
  const int lane = threadIdx.x & 31, wave = threadIdx.x >> 5;
  #pragma unroll
  for (int o = 16; o >= 1; o >>= 1) v = opApply(v, __shfl_xor(v, o, 32), op);
  if (lane == 0) sred[wave] = v;
  __syncthreads();
  const int nw = blockDim.x >> 5;
  float acc = sred[0];
  for (int w = 1; w < nw; ++w) acc = opApply(acc, sred[w], op);
  __syncthreads();
  return acc;
}

// ---------------- fq16: global min/max -> scale/zp ----------------
__global__ void minmax_partial(const float* __restrict__ x, long n,
                               float* __restrict__ pmin, float* __restrict__ pmax) {
  __shared__ float sred[8];
  float lmin = 3.4e38f, lmax = -3.4e38f;
  for (long i = (long)blockIdx.x * blockDim.x + threadIdx.x; i < n;
       i += (long)gridDim.x * blockDim.x) {
    const float v = x[i];
    lmin = fminf(lmin, v);
    lmax = fmaxf(lmax, v);
  }
  const float mn = blockReduce(lmin, 0, sred);
  const float mx = blockReduce(lmax, 1, sred);
  if (threadIdx.x == 0) { pmin[blockIdx.x] = mn; pmax[blockIdx.x] = mx; }
}

__global__ void minmax_final(const float* __restrict__ pmin, const float* __restrict__ pmax,
                             int nb, float* __restrict__ par) {
  __shared__ float sred[8];
  float lmin = 3.4e38f, lmax = -3.4e38f;
  for (int i = threadIdx.x; i < nb; i += blockDim.x) {
    lmin = fminf(lmin, pmin[i]);
    lmax = fmaxf(lmax, pmax[i]);
  }
  float mn = blockReduce(lmin, 0, sred);
  float mx = blockReduce(lmax, 1, sred);
  if (threadIdx.x == 0) {
    mn = fminf(mn, 0.f); mx = fmaxf(mx, 0.f);
    const float scale = fmaxf((mx - mn) / 65535.0f, EPS_);
    par[0] = scale;
    par[1] = rintf(-mn / scale);
  }
}

__global__ void fq16_cvt(const float* __restrict__ x, _Float16* __restrict__ y, long n,
                         const float* __restrict__ par) {
  const float scale = par[0], zp = par[1];
  for (long i = (long)blockIdx.x * blockDim.x + threadIdx.x; i < n;
       i += (long)gridDim.x * blockDim.x) {
    float q = rintf(x[i] / scale) + zp;
    q = fminf(fmaxf(q, 0.0f), 65535.0f);
    y[i] = (_Float16)((q - zp) * scale);
  }
}

// ---------------- fq8_sym: absmax -> int8 grid snap ----------------
__global__ void absmax_partial(const _Float16* __restrict__ x, long n, float* __restrict__ pabs) {
  __shared__ float sred[8];
  float lmax = 0.0f;
  for (long i = (long)blockIdx.x * blockDim.x + threadIdx.x; i < n;
       i += (long)gridDim.x * blockDim.x)
    lmax = fmaxf(lmax, fabsf((float)x[i]));
  const float mx = blockReduce(lmax, 1, sred);
  if (threadIdx.x == 0) pabs[blockIdx.x] = mx;
}

__global__ void absmax_final(const float* __restrict__ pabs, int nb,
                             float* __restrict__ par, int idx) {
  __shared__ float sred[8];
  float lmax = 0.0f;
  for (int i = threadIdx.x; i < nb; i += blockDim.x) lmax = fmaxf(lmax, pabs[i]);
  const float mx = blockReduce(lmax, 1, sred);
  if (threadIdx.x == 0) par[idx] = fmaxf(mx / 127.0f, 1e-12f);
}

__global__ void snap8(_Float16* __restrict__ x, long n, const float* __restrict__ par, int idx) {
  const float s = par[idx];
  for (long i = (long)blockIdx.x * blockDim.x + threadIdx.x; i < n;
       i += (long)gridDim.x * blockDim.x) {
    const float q = fminf(fmaxf(rintf((float)x[i] / s), -128.0f), 127.0f);
    x[i] = (_Float16)(q * s);
  }
}

// ---------------- LPBQ weight fake-quant (block=32, qmax=7) -> f16 ----------------
__global__ void lpbq_w(const float* __restrict__ W, _Float16* __restrict__ Wq, long nblocks) {
  const long b = (long)blockIdx.x * blockDim.x + threadIdx.x;
  if (b >= nblocks) return;
  const float* w = W + b * 32;
  float amax = 0.0f;
  #pragma unroll
  for (int i = 0; i < 32; ++i) amax = fmaxf(amax, fabsf(w[i]));
  const float s = fmaxf(amax / 7.0f, 1e-12f);
  #pragma unroll
  for (int i = 0; i < 32; ++i) {
    const float q = fminf(fmaxf(rintf(w[i] / s), -8.0f), 7.0f);
    Wq[b * 32 + i] = (_Float16)(q * s);
  }
}

// ---------------- fused RMSNorm + RoPE (one wave per (b,s,h) row of 128) ----------------
__global__ __launch_bounds__(128) void rms_rope(const float* __restrict__ X,
                                                const float* __restrict__ nw,
                                                const float* __restrict__ cosb,
                                                const float* __restrict__ sinb,
                                                _Float16* __restrict__ out,
                                                int H, int rowStride) {
  const int lane = threadIdx.x & 31;
  const int gw = blockIdx.x * (blockDim.x >> 5) + (threadIdx.x >> 5);
  const int h = gw % H;
  const int s = (gw / H) % S_;
  const int b = gw / (H * S_);
  const float* x = X + (long)(b * S_ + s) * rowStride + h * D_;
  const int d0 = lane * 4;
  float x0 = x[d0], x1 = x[d0 + 1], x2 = x[d0 + 2], x3 = x[d0 + 3];
  float ss = x0 * x0 + x1 * x1 + x2 * x2 + x3 * x3;
  #pragma unroll
  for (int o = 16; o >= 1; o >>= 1) ss += __shfl_xor(ss, o, 32);
  const float r = rsqrtf(ss / (float)D_ + 1e-6f);
  const float n0 = x0 * r * nw[d0],     n1 = x1 * r * nw[d0 + 1];
  const float n2 = x2 * r * nw[d0 + 2], n3 = x3 * r * nw[d0 + 3];
  // rotate-half partner lives 64 elements away -> lane ^ 16 under wave32
  const float p0 = __shfl_xor(n0, 16, 32), p1 = __shfl_xor(n1, 16, 32);
  const float p2 = __shfl_xor(n2, 16, 32), p3 = __shfl_xor(n3, 16, 32);
  const float sgn = (lane < 16) ? -1.0f : 1.0f;
  const float* cp = cosb + (long)(b * S_ + s) * D_ + d0;
  const float* sp = sinb + (long)(b * S_ + s) * D_ + d0;
  _Float16* y = out + ((long)(b * H + h) * S_ + s) * D_ + d0;
  y[0] = (_Float16)(n0 * cp[0] + sgn * p0 * sp[0]);
  y[1] = (_Float16)(n1 * cp[1] + sgn * p1 * sp[1]);
  y[2] = (_Float16)(n2 * cp[2] + sgn * p2 * sp[2]);
  y[3] = (_Float16)(n3 * cp[3] + sgn * p3 * sp[3]);
}

// ---------------- V pack: [b,s,kv,d] f32 -> vt[b,kv,d,s] f16 (transposed for AV GEMM) ----
__global__ void v_pack(const float* __restrict__ v0, _Float16* __restrict__ vt) {
  const long i = (long)blockIdx.x * blockDim.x + threadIdx.x; // over B*NKV*D*S
  const int s  = (int)(i & (S_ - 1));
  const int d  = (int)((i >> 10) & (D_ - 1));
  const int kv = (int)((i >> 17) & (NKV_ - 1));
  const int b  = (int)(i >> 20);
  vt[i] = (_Float16)v0[(long)(b * S_ + s) * (NKV_ * D_) + kv * D_ + d];
}

// ---------------- masked softmax with min-fill (one block per score row) ----------------
__global__ __launch_bounds__(256) void softmax_mask(float* __restrict__ attn,
                                                    const int* __restrict__ mask) {
  __shared__ float sred[8];
  const long row = blockIdx.x;                 // (b*NH + h)*S + srow
  const int srow = (int)(row & (S_ - 1));
  float* p = attn + row * S_;
  const int* mrow = mask + (long)srow * S_;
  const float scaling = 0.088388347648318447f; // 128^-0.5
  const int t0 = threadIdx.x * 4;
  float v[4];
  float lmin = 3.4e38f;
  #pragma unroll
  for (int i = 0; i < 4; ++i) { v[i] = p[t0 + i] * scaling; lmin = fminf(lmin, v[i]); }
  const float rmin = blockReduce(lmin, 0, sred);
  const float fill = rmin - 20.0f;
  float lmax = -3.4e38f;
  #pragma unroll
  for (int i = 0; i < 4; ++i) {
    if (mrow[t0 + i] != 0) v[i] = fill;
    lmax = fmaxf(lmax, v[i]);
  }
  const float rmax = blockReduce(lmax, 1, sred);
  float lsum = 0.0f;
  #pragma unroll
  for (int i = 0; i < 4; ++i) { v[i] = __expf(v[i] - rmax); lsum += v[i]; }
  const float inv = 1.0f / blockReduce(lsum, 2, sred);
  #pragma unroll
  for (int i = 0; i < 4; ++i) p[t0 + i] = v[i] * inv;
}

// ---------------- WMMA GEMM: C[M,N] = A[M,K] * B[N,K]^T ----------------
// f16-A path: double-buffered LDS filled with GLOBAL_LOAD_ASYNC_TO_LDS_B128 (ASYNCcnt),
//             next tile's DMA overlaps current tile's WMMA; one barrier per K step.
// f32-A path (softmax probs): synchronous convert-on-fill staging.
// Batch decomposition: z -> (b = z/NHp, h = z%NHp); GQA via h/gqaDiv on B.
template <typename AT, typename CT>
__global__ __launch_bounds__(256) void gemm_wmma(const AT* __restrict__ A,
                                                 const _Float16* __restrict__ Bw,
                                                 CT* __restrict__ C,
                                                 int M, int N, int K,
                                                 int lda, int ldb, int ldc,
                                                 long aB, long aH, long bB, long bH,
                                                 long cB, long cH, int NHp, int gqaDiv) {
  __shared__ _Float16 lA[2][BM * KTP];
  __shared__ _Float16 lB[2][BN * KTP];
  const int z  = blockIdx.z;
  const int bb = z / NHp, hh = z % NHp;
  const AT* Ab = A + bb * aB + hh * aH;
  const _Float16* Bb = Bw + bb * bB + (long)(hh / gqaDiv) * bH;
  CT* Cb = C + bb * cB + hh * cH;

  const int m0 = blockIdx.y * BM;
  const int n0 = blockIdx.x * BN;
  const int tid = threadIdx.x;
  const int lane = tid & 31, wave = tid >> 5;
  const int wm = wave >> 1, wn = wave & 1;        // 4x2 wave grid, 32x32 per wave
  const int arow = tid >> 1, aseg = tid & 1;      // A fill: 128 rows x 2 segs of 16 halves
  const int brow = tid >> 2, bseg = tid & 3;      // B fill: 64 rows x 4 segs of 8 halves
  const int ln = lane & 15, lk = (lane >> 4) * 16; // fragment lane mapping (contiguous k = 16*(lane/16) + s)
  v8f c[2][2] = {};

  const int steps = K / KT;

  if constexpr (sizeof(AT) == 2) {
    // -------- async DMA double-buffered pipeline --------
    auto issue = [&](int buf, int kt) {
      const int kk = kt * KT;
      const _Float16* ga = (const _Float16*)Ab + (long)(m0 + arow) * lda + kk + aseg * 16;
      const unsigned la = (unsigned)(size_t)&lA[buf][arow * KTP + aseg * 16];
      asm volatile("global_load_async_to_lds_b128 %0, %1, off"
                   :: "v"(la), "v"((unsigned long long)(size_t)ga) : "memory");
      asm volatile("global_load_async_to_lds_b128 %0, %1, off"
                   :: "v"(la + 16u), "v"((unsigned long long)(size_t)(ga + 8)) : "memory");
      const _Float16* gb = Bb + (long)(n0 + brow) * ldb + kk + bseg * 8;
      const unsigned lb = (unsigned)(size_t)&lB[buf][brow * KTP + bseg * 8];
      asm volatile("global_load_async_to_lds_b128 %0, %1, off"
                   :: "v"(lb), "v"((unsigned long long)(size_t)gb) : "memory");
    };
    issue(0, 0);
    asm volatile("s_wait_asynccnt 0x0" ::: "memory");
    __syncthreads();
    for (int kt = 0; kt < steps; ++kt) {
      const int buf = kt & 1;
      if (kt + 1 < steps) issue(buf ^ 1, kt + 1); // DMA of next tile overlaps this tile's math
      if (kt + 2 < steps) {                       // far prefetch into L2
        __builtin_prefetch((const _Float16*)Ab + (long)(m0 + arow) * lda + (kt + 2) * KT, 0, 1);
        __builtin_prefetch(Bb + (long)(n0 + brow) * ldb + (kt + 2) * KT, 0, 1);
      }
      v16h af[2], bf[2];
      #pragma unroll
      for (int i = 0; i < 2; ++i)
        af[i] = *(const v16h*)&lA[buf][(wm * 32 + i * 16 + ln) * KTP + lk];
      #pragma unroll
      for (int j = 0; j < 2; ++j)
        bf[j] = *(const v16h*)&lB[buf][(wn * 32 + j * 16 + ln) * KTP + lk];
      #pragma unroll
      for (int i = 0; i < 2; ++i)
        #pragma unroll
        for (int j = 0; j < 2; ++j)
          c[i][j] = __builtin_amdgcn_wmma_f32_16x16x32_f16(
              false, af[i], false, bf[j], (short)0, c[i][j], false, false);
      if (kt + 1 < steps)
        asm volatile("s_wait_asynccnt 0x0" ::: "memory");
      __syncthreads();
    }
  } else {
    // -------- synchronous staging with f32 -> f16 convert on fill --------
    for (int kt = 0; kt < steps; ++kt) {
      const int kk = kt * KT;
      {
        const float* src = (const float*)Ab + (long)(m0 + arow) * lda + kk + aseg * 16;
        v8h h0, h1;
        #pragma unroll
        for (int i = 0; i < 8; ++i) { h0[i] = (_Float16)src[i]; h1[i] = (_Float16)src[8 + i]; }
        *(v8h*)&lA[0][arow * KTP + aseg * 16 + 0] = h0;
        *(v8h*)&lA[0][arow * KTP + aseg * 16 + 8] = h1;
      }
      {
        const v8h b0 = *(const v8h*)(Bb + (long)(n0 + brow) * ldb + kk + bseg * 8);
        *(v8h*)&lB[0][brow * KTP + bseg * 8] = b0;
      }
      __syncthreads();
      if (kt + 1 < steps) {
        __builtin_prefetch((const float*)Ab + (long)(m0 + arow) * lda + kk + KT + aseg * 16, 0, 1);
        __builtin_prefetch(Bb + (long)(n0 + brow) * ldb + kk + KT + bseg * 8, 0, 1);
      }
      v16h af[2], bf[2];
      #pragma unroll
      for (int i = 0; i < 2; ++i)
        af[i] = *(const v16h*)&lA[0][(wm * 32 + i * 16 + ln) * KTP + lk];
      #pragma unroll
      for (int j = 0; j < 2; ++j)
        bf[j] = *(const v16h*)&lB[0][(wn * 32 + j * 16 + ln) * KTP + lk];
      #pragma unroll
      for (int i = 0; i < 2; ++i)
        #pragma unroll
        for (int j = 0; j < 2; ++j)
          c[i][j] = __builtin_amdgcn_wmma_f32_16x16x32_f16(
              false, af[i], false, bf[j], (short)0, c[i][j], false, false);
      __syncthreads();
    }
  }

  // ---- store C (ISA 16x16 f32 layout: lane<16 -> M=vgpr, lane>=16 -> M=8+vgpr) ----
  const int cn = lane & 15;
  const int mb = (lane >> 4) * 8;
  #pragma unroll
  for (int i = 0; i < 2; ++i)
    #pragma unroll
    for (int j = 0; j < 2; ++j) {
      const int gm = m0 + wm * 32 + i * 16 + mb;
      const int gn = n0 + wn * 32 + j * 16 + cn;
      #pragma unroll
      for (int r = 0; r < 8; ++r)
        Cb[(long)(gm + r) * ldc + gn] = (CT)c[i][j][r];
    }
}

// ---------------- host-side orchestration ----------------
extern "C" void kernel_launch(void* const* d_in, const int* in_sizes, int n_in,
                              void* d_out, int out_size, void* d_ws, size_t ws_size,
                              hipStream_t stream) {
  (void)in_sizes; (void)n_in; (void)out_size;
  const float* hs   = (const float*)d_in[0];
  const float* cosb = (const float*)d_in[1];
  const float* sinb = (const float*)d_in[2];
  const int*   mask = (const int*)d_in[3];
  const float* Wq   = (const float*)d_in[4];
  const float* Wk   = (const float*)d_in[5];
  const float* Wv   = (const float*)d_in[6];
  const float* Wo   = (const float*)d_in[7];
  const float* qnw  = (const float*)d_in[8];
  const float* knw  = (const float*)d_in[9];

  char* ws = (char*)d_ws;
  size_t off = 0;
  auto carve = [&](size_t bytes) -> char* {
    char* p = ws + off;
    off = (off + bytes + 255) & ~(size_t)255;
    return p;
  };
  float*    par  = (float*)carve(64 * sizeof(float)); // [0..1]=hs scale/zp, [2]=k8, [3]=v8
  float*    pmin = (float*)carve(1024 * sizeof(float));
  float*    pmax = (float*)carve(1024 * sizeof(float));
  _Float16* hsq  = (_Float16*)carve((size_t)B_ * S_ * HID_ * 2);
  _Float16* wqq  = (_Float16*)carve((size_t)HID_ * HID_ * 2);
  _Float16* wkq  = (_Float16*)carve((size_t)NKV_ * D_ * HID_ * 2);
  _Float16* wvq  = (_Float16*)carve((size_t)NKV_ * D_ * HID_ * 2);
  _Float16* woq  = (_Float16*)carve((size_t)HID_ * HID_ * 2);
  float*    q0   = (float*)carve((size_t)B_ * S_ * NH_ * D_ * 4);
  float*    k0   = (float*)carve((size_t)B_ * S_ * NKV_ * D_ * 4);
  float*    v0   = (float*)carve((size_t)B_ * S_ * NKV_ * D_ * 4);
  _Float16* qh   = (_Float16*)carve((size_t)B_ * NH_ * S_ * D_ * 2);
  _Float16* kh   = (_Float16*)carve((size_t)B_ * NKV_ * S_ * D_ * 2);
  _Float16* vt   = (_Float16*)carve((size_t)B_ * NKV_ * D_ * S_ * 2);
  _Float16* av   = (_Float16*)carve((size_t)B_ * S_ * HID_ * 2);
  if (off > ws_size) return;

  float* outp  = (float*)d_out;
  float* attnp = outp + (size_t)B_ * S_ * HID_;

  const long nHS = (long)B_ * S_ * HID_;
  const long nKV = (long)B_ * NKV_ * S_ * D_;

  // fq16(hidden) -> f16
  minmax_partial<<<1024, 256, 0, stream>>>(hs, nHS, pmin, pmax);
  minmax_final<<<1, 256, 0, stream>>>(pmin, pmax, 1024, par);
  fq16_cvt<<<4096, 256, 0, stream>>>(hs, hsq, nHS, par);

  // LPBQ fake-quant weights -> f16
  lpbq_w<<<(HID_ * HID_ / 32 + 255) / 256, 256, 0, stream>>>(Wq, wqq, (long)HID_ * HID_ / 32);
  lpbq_w<<<(NKV_ * D_ * HID_ / 32 + 255) / 256, 256, 0, stream>>>(Wk, wkq, (long)NKV_ * D_ * HID_ / 32);
  lpbq_w<<<(NKV_ * D_ * HID_ / 32 + 255) / 256, 256, 0, stream>>>(Wv, wvq, (long)NKV_ * D_ * HID_ / 32);
  lpbq_w<<<(HID_ * HID_ / 32 + 255) / 256, 256, 0, stream>>>(Wo, woq, (long)HID_ * HID_ / 32);

  const dim3 blk(256);
  // QKV projections (f16 WMMA, async-LDS pipeline, f32 out)
  gemm_wmma<_Float16, float><<<dim3(HID_ / BN, (B_ * S_) / BM, 1), blk, 0, stream>>>(
      hsq, wqq, q0, B_ * S_, HID_, HID_, HID_, HID_, HID_, 0, 0, 0, 0, 0, 0, 1, 1);
  gemm_wmma<_Float16, float><<<dim3((NKV_ * D_) / BN, (B_ * S_) / BM, 1), blk, 0, stream>>>(
      hsq, wkq, k0, B_ * S_, NKV_ * D_, HID_, HID_, HID_, NKV_ * D_, 0, 0, 0, 0, 0, 0, 1, 1);
  gemm_wmma<_Float16, float><<<dim3((NKV_ * D_) / BN, (B_ * S_) / BM, 1), blk, 0, stream>>>(
      hsq, wvq, v0, B_ * S_, NKV_ * D_, HID_, HID_, HID_, NKV_ * D_, 0, 0, 0, 0, 0, 0, 1, 1);

  // RMSNorm + RoPE (q, k), V transpose/pack
  rms_rope<<<(B_ * S_ * NH_) / 4, 128, 0, stream>>>(q0, qnw, cosb, sinb, qh, NH_, HID_);
  rms_rope<<<(B_ * S_ * NKV_) / 4, 128, 0, stream>>>(k0, knw, cosb, sinb, kh, NKV_, NKV_ * D_);
  v_pack<<<(B_ * NKV_ * D_ * S_) / 256, 256, 0, stream>>>(v0, vt);

  // fq8_sym(k), fq8_sym(v): global absmax then snap to int8 grid
  absmax_partial<<<1024, 256, 0, stream>>>(kh, nKV, pmin);
  absmax_final<<<1, 256, 0, stream>>>(pmin, 1024, par, 2);
  snap8<<<2048, 256, 0, stream>>>(kh, nKV, par, 2);
  absmax_partial<<<1024, 256, 0, stream>>>(vt, nKV, pmin);
  absmax_final<<<1, 256, 0, stream>>>(pmin, 1024, par, 3);
  snap8<<<2048, 256, 0, stream>>>(vt, nKV, par, 3);

  // scores = Q K^T  (written straight into d_out attn region; it lives in L2 afterwards)
  gemm_wmma<_Float16, float><<<dim3(S_ / BN, S_ / BM, B_ * NH_), blk, 0, stream>>>(
      qh, kh, attnp, S_, S_, D_, D_, D_, S_,
      (long)NH_ * S_ * D_, (long)S_ * D_, (long)NKV_ * S_ * D_, (long)S_ * D_,
      (long)NH_ * S_ * S_, (long)S_ * S_, NH_, 2);

  // scale + min-fill mask + softmax, in place (this IS the attn output)
  softmax_mask<<<B_ * NH_ * S_, 256, 0, stream>>>(attnp, mask);

  // out = P V   (A is f32 probs, converted to f16 during LDS staging; write [b,s,h*D+d])
  gemm_wmma<float, _Float16><<<dim3(D_ / BN, S_ / BM, B_ * NH_), blk, 0, stream>>>(
      attnp, vt, av, S_, D_, S_, S_, S_, HID_,
      (long)NH_ * S_ * S_, (long)S_ * S_, (long)NKV_ * D_ * S_, (long)D_ * S_,
      (long)S_ * HID_, (long)D_, NH_, 2);

  // final output projection -> d_out (f32)
  gemm_wmma<_Float16, float><<<dim3(HID_ / BN, (B_ * S_) / BM, 1), blk, 0, stream>>>(
      av, woq, outp, B_ * S_, HID_, HID_, HID_, HID_, HID_, 0, 0, 0, 0, 0, 0, 1, 1);
}
